// TokenOTBlock_54056458387746
// MI455X (gfx1250) — matrix operand order repaired and checked by reading, not compiled, not run on hardware
//
#include <hip/hip_runtime.h>

typedef __attribute__((ext_vector_type(8)))  float        v8f;
typedef __attribute__((ext_vector_type(16))) __bf16       v16bf;
typedef __attribute__((ext_vector_type(4)))  unsigned int u32x4;
typedef __attribute__((ext_vector_type(8)))  int          i32x8;
typedef __attribute__((ext_vector_type(4)))  int          i32x4;
typedef __attribute__((ext_vector_type(4)))  float        f32x4;

#define WMMA_BF16(a, b, c) \
  __builtin_amdgcn_wmma_f32_16x16x32_bf16(false, (a), false, (b), (short)0, (c), false, false)

// Fragment = 32 bytes assembled from two 16-byte chunks (contiguous K runs).
union Frag { u32x4 q[2]; v16bf v; };

__device__ __forceinline__ unsigned short f32_to_bf16_bits(float x) {
  unsigned int u = __builtin_bit_cast(unsigned int, x);
  u = u + 0x7fffu + ((u >> 16) & 1u);   // round-to-nearest-even
  return (unsigned short)(u >> 16);
}
__device__ __forceinline__ __bf16 bfbits(unsigned short u) {
  return __builtin_bit_cast(__bf16, u);
}
__device__ __forceinline__ __bf16 to_bf(float x) { return bfbits(f32_to_bf16_bits(x)); }

#define NEGV (-1e9f)

// ---------------------------------------------------------------- utilities
__global__ void cvt_kernel(const float* __restrict__ x, unsigned short* __restrict__ y, int n) {
  int i = blockIdx.x * blockDim.x + threadIdx.x;
  if (i < n) y[i] = f32_to_bf16_bits(x[i]);
}

// f32 (K x Nw) row-major -> bf16 (Nw x K) row-major (weight transpose), 32x32 LDS tiles
__global__ __launch_bounds__(256) void cvt_t_kernel(const float* __restrict__ in,
                                                    unsigned short* __restrict__ out,
                                                    int K, int Nw) {
  __shared__ float tile[32][33];
  int k0 = blockIdx.y * 32, n0 = blockIdx.x * 32;
  int tx = threadIdx.x & 31, ty = threadIdx.x >> 5;
  for (int i = ty; i < 32; i += 8) tile[i][tx] = in[(size_t)(k0 + i) * Nw + n0 + tx];
  __syncthreads();
  for (int i = ty; i < 32; i += 8)
    out[(size_t)(n0 + i) * K + k0 + tx] = f32_to_bf16_bits(tile[tx][i]);
}

__global__ void zero_kernel(float* __restrict__ p, int n) {
  int i = blockIdx.x * blockDim.x + threadIdx.x;
  if (i < n) p[i] = 0.f;
}

__global__ void copy_kernel(const float* __restrict__ s, float* __restrict__ d, int n) {
  int i = blockIdx.x * blockDim.x + threadIdx.x;
  if (i < n) d[i] = s[i];
}

// ------------------------------------------------------------- layernorm -> bf16
__global__ __launch_bounds__(256) void ln_bf16_kernel(
    const float* __restrict__ x, const float* __restrict__ g, const float* __restrict__ b,
    unsigned short* __restrict__ y, int D) {
  __shared__ float red[256];
  int n = blockIdx.x, tid = threadIdx.x;
  const float* xr = x + (size_t)n * D;
  float s = 0.f;
  for (int d = tid; d < D; d += 256) s += xr[d];
  red[tid] = s; __syncthreads();
  for (int st = 128; st > 0; st >>= 1) { if (tid < st) red[tid] += red[tid + st]; __syncthreads(); }
  float mu = red[0] / (float)D;
  __syncthreads();
  float v = 0.f;
  for (int d = tid; d < D; d += 256) { float t = xr[d] - mu; v += t * t; }
  red[tid] = v; __syncthreads();
  for (int st = 128; st > 0; st >>= 1) { if (tid < st) red[tid] += red[tid + st]; __syncthreads(); }
  float rs = rsqrtf(red[0] / (float)D + 1e-5f);
  for (int d = tid; d < D; d += 256)
    y[(size_t)n * D + d] = f32_to_bf16_bits((xr[d] - mu) * rs * g[d] + b[d]);
}

// ------------------------------------------------------------- bf16 WMMA GEMM
// C[M,Nw] = A[M,K] (bf16 row-major) x W^T[Nw,K] (bf16 row-major), fp32 out.
// A tile staged via vector global loads; W^T tile staged via the Tensor Data Mover.
// Block = 128 threads (4 waves), block tile 64x64, K step 32.
__global__ __launch_bounds__(128) void gemm_bf16_kernel(
    const unsigned short* __restrict__ A, const unsigned short* __restrict__ WT,
    float* __restrict__ C, int M, int Nw, int K) {
  __shared__ unsigned short As[64 * 32];   // 64 rows (m) x 32 k
  __shared__ unsigned short Bs[64 * 32];   // 64 rows (n) x 32 k
  const int tid = threadIdx.x;
  const int wave = tid >> 5, lane = tid & 31;
  const int lm = lane & 15, lh = lane >> 4;
  const int m0 = blockIdx.y * 64, n0 = blockIdx.x * 64;

  v8f acc[4] = {v8f{}, v8f{}, v8f{}, v8f{}};

  for (int kb = 0; kb < K; kb += 32) {
    // --- A tile: 1024 dwords cooperative load
#pragma unroll
    for (int i = 0; i < 8; ++i) {
      int u = tid + i * 128;
      int row = u >> 4, kk = u & 15;
      ((unsigned int*)As)[u] = *(const unsigned int*)(A + (size_t)(m0 + row) * K + kb + 2 * kk);
    }
    if (kb + 32 < K) {
      int row = tid >> 4, kk = tid & 15;
      __builtin_prefetch(A + (size_t)(m0 + row) * K + (kb + 32) + 2 * kk, 0, 1);
    }
    // --- B tile: TDM async tensor load (2D: 32 elems x 64 rows of W^T), wave 0 issues
    if (tid < 32) {
      unsigned lds_b = (unsigned)(unsigned long long)(void*)&Bs[0];
      unsigned long long ga = (unsigned long long)(const void*)(WT + (size_t)n0 * K + kb);
      u32x4 g0;
      g0[0] = 1u;                                           // count=1 (valid user D#)
      g0[1] = lds_b;                                        // lds_addr
      g0[2] = (unsigned)(ga & 0xffffffffu);                 // global_addr[31:0]
      g0[3] = (unsigned)((ga >> 32) & 0x01ffffffu) | 0x80000000u; // addr[56:32] | type=2
      i32x8 g1;
      g1[0] = 0x10000;                                      // data_size=1 (2 bytes)
      g1[1] = (int)((K & 0xffff) << 16);                    // tensor_dim0 lo16
      g1[2] = (int)(((unsigned)K >> 16) | ((Nw & 0xffff) << 16)); // dim0 hi | dim1 lo
      g1[3] = (int)(((unsigned)Nw >> 16) | (32u << 16));    // dim1 hi | tile_dim0=32
      g1[4] = 64;                                           // tile_dim1=64
      g1[5] = K;                                            // tensor_dim0_stride lo32
      g1[6] = 0;
      g1[7] = 0;
      i32x4 z4 = {0, 0, 0, 0};
      i32x8 z8 = {0, 0, 0, 0, 0, 0, 0, 0};
      __builtin_amdgcn_tensor_load_to_lds(g0, g1, z4, z4, z8, 0);
    }
    __builtin_amdgcn_s_wait_tensorcnt(0);
    __syncthreads();

    // A fragment: two contiguous 16B K-runs -> 2x ds_load_b128
    Frag fa;
    {
      const int abase = (wave * 16 + lm) * 32;
      fa.q[0] = *(const u32x4*)&As[abase + lh * 8];
      fa.q[1] = *(const u32x4*)&As[abase + 16 + lh * 8];
    }
#pragma unroll
    for (int nt = 0; nt < 4; ++nt) {
      Frag fb;
      const int nbase = (nt * 16 + lm) * 32;
      fb.q[0] = *(const u32x4*)&Bs[nbase + lh * 8];
      fb.q[1] = *(const u32x4*)&Bs[nbase + 16 + lh * 8];
      acc[nt] = WMMA_BF16(fa.v, fb.v, acc[nt]);
    }
    __syncthreads();
  }
#pragma unroll
  for (int nt = 0; nt < 4; ++nt)
#pragma unroll
    for (int r = 0; r < 8; ++r)
      C[(size_t)(m0 + wave * 16 + r + 8 * lh) * Nw + n0 + nt * 16 + lm] = acc[nt][r];
}

// ------------------------------------------- (N,D) fp32 -> head-major bf16
// mode 0: (H,N,DH)   mode 1: (H,N,DH) + L2-normalize over DH   mode 2: (H,DH,N) (transposed)
__global__ __launch_bounds__(32) void repack_kernel(
    const float* __restrict__ C, unsigned short* __restrict__ out, int N, int mode) {
  int n = blockIdx.x, h = blockIdx.y, dh = threadIdx.x;
  float v = C[(size_t)n * 512 + h * 32 + dh];
  if (mode == 1) {
    float ss = v * v;
#pragma unroll
    for (int o = 16; o > 0; o >>= 1) ss += __shfl_xor(ss, o, 32);
    float nrm = fmaxf(sqrtf(ss), 1e-12f);
    v /= nrm;
  }
  if (mode == 2)
    out[((size_t)h * 32 + dh) * N + n] = f32_to_bf16_bits(v);
  else
    out[((size_t)h * N + n) * 32 + dh] = f32_to_bf16_bits(v);
}

// ------------------------------------------------------------- flash attention (WMMA x4 / 32 keys)
// One wave per (head, 16-query tile). S^T = K*Q^T (2 tiles), online softmax, O^T = V^T*P^T (K=32).
__global__ __launch_bounds__(32) void attn_kernel(
    const unsigned short* __restrict__ Qh, const unsigned short* __restrict__ Kh,
    const unsigned short* __restrict__ Vt, const int* __restrict__ pos_bins,
    const float* __restrict__ pos_w, const float* __restrict__ mask,
    float* __restrict__ att, int N) {
  const int h = blockIdx.y;
  const int qb = blockIdx.x * 16;
  const int lane = threadIdx.x;
  const int q = lane & 15, lh = lane >> 4;
  const float scale = 0.17677669529663687f; // 1/sqrt(32)

  // B fragment = Q^T (dh x query): two 16B runs per lane (dh lh*8.., 16+lh*8..)
  Frag bq;
  {
    const unsigned short* Qp = Qh + ((size_t)h * N + qb + q) * 32;
    bq.q[0] = *(const u32x4*)(Qp + lh * 8);
    bq.q[1] = *(const u32x4*)(Qp + 16 + lh * 8);
  }
  const int* brow = pos_bins + (size_t)(qb + q) * N;

  float m_run = -3.0e38f, s_run = 0.f;
  v8f o0 = {}, o1 = {};

  for (int kb = 0; kb < N; kb += 32) {
    // --- two S^T tiles (keys kb..kb+15, kb+16..kb+31)
    Frag fk0, fk1;
    {
      const unsigned short* kp0 = Kh + ((size_t)h * N + kb + q) * 32;
      const unsigned short* kp1 = kp0 + 16 * 32;
      fk0.q[0] = *(const u32x4*)(kp0 + lh * 8);
      fk0.q[1] = *(const u32x4*)(kp0 + 16 + lh * 8);
      fk1.q[0] = *(const u32x4*)(kp1 + lh * 8);
      fk1.q[1] = *(const u32x4*)(kp1 + 16 + lh * 8);
    }
    v8f z0 = {}, z1 = {};
    v8f s0 = WMMA_BF16(fk0.v, bq.v, z0);
    v8f s1 = WMMA_BF16(fk1.v, bq.v, z1);

    // bias + mask, vector row loads
    i32x4 bA0 = *(const i32x4*)(brow + kb + 8 * lh);
    i32x4 bA1 = *(const i32x4*)(brow + kb + 8 * lh + 4);
    i32x4 bB0 = *(const i32x4*)(brow + kb + 16 + 8 * lh);
    i32x4 bB1 = *(const i32x4*)(brow + kb + 16 + 8 * lh + 4);
    f32x4 mA0 = *(const f32x4*)(mask + kb + 8 * lh);
    f32x4 mA1 = *(const f32x4*)(mask + kb + 8 * lh + 4);
    f32x4 mB0 = *(const f32x4*)(mask + kb + 16 + 8 * lh);
    f32x4 mB1 = *(const f32x4*)(mask + kb + 16 + 8 * lh + 4);

    float p0[8], p1[8];
    float mt = -3.0e38f;
#pragma unroll
    for (int r = 0; r < 8; ++r) {
      int bin0 = (r < 4) ? bA0[r & 3] : bA1[r & 3];
      int bin1 = (r < 4) ? bB0[r & 3] : bB1[r & 3];
      float mk0 = (r < 4) ? mA0[r & 3] : mA1[r & 3];
      float mk1 = (r < 4) ? mB0[r & 3] : mB1[r & 3];
      float l0 = s0[r] * scale + pos_w[(size_t)bin0 * 16 + h];
      float l1 = s1[r] * scale + pos_w[(size_t)bin1 * 16 + h];
      l0 = (mk0 > 0.f) ? l0 : NEGV;
      l1 = (mk1 > 0.f) ? l1 : NEGV;
      p0[r] = l0; p1[r] = l1;
      mt = fmaxf(mt, fmaxf(l0, l1));
    }
    mt = fmaxf(mt, __shfl_xor(mt, 16, 32));
    float m_new = fmaxf(m_run, mt);
    float resc = __expf(m_run - m_new);
    float ps = 0.f;
#pragma unroll
    for (int r = 0; r < 8; ++r) {
      p0[r] = __expf(p0[r] - m_new); ps += p0[r];
      p1[r] = __expf(p1[r] - m_new); ps += p1[r];
    }
    ps += __shfl_xor(ps, 16, 32);
    s_run = s_run * resc + ps;
    m_run = m_new;
    o0 *= resc;
    o1 *= resc;

    // B fragment = P^T (32 keys x 16 queries); partner-half values via shfl_xor(16)
    float po0[8], po1[8];
#pragma unroll
    for (int r = 0; r < 8; ++r) {
      po0[r] = __shfl_xor(p0[r], 16, 32);
      po1[r] = __shfl_xor(p1[r], 16, 32);
    }
    v16bf bp;
#pragma unroll
    for (int j = 0; j < 8; ++j) {
      float lo, hi;
      if (j < 4) {            // K = lh*16 + 2j -> keys 0..7 (h0) / 16..23 (h1)
        lo = lh ? po1[2 * j]     : p0[2 * j];
        hi = lh ? po1[2 * j + 1] : p0[2 * j + 1];
      } else {                // keys 8..15 (h0) / 24..31 (h1)
        int jj = 2 * (j - 4);
        lo = lh ? p1[jj]     : po0[jj];
        hi = lh ? p1[jj + 1] : po0[jj + 1];
      }
      bp[2 * j]     = to_bf(lo);
      bp[2 * j + 1] = to_bf(hi);
    }

    // A fragments = V^T tiles (dh x 32 keys): contiguous runs in (H,DH,N) layout
    Frag fa0, fa1;
    {
      const unsigned short* vp0 = Vt + ((size_t)h * 32 + q) * N + kb;
      const unsigned short* vp1 = Vt + ((size_t)h * 32 + 16 + q) * N + kb;
      fa0.q[0] = *(const u32x4*)(vp0 + lh * 16);
      fa0.q[1] = *(const u32x4*)(vp0 + lh * 16 + 8);
      fa1.q[0] = *(const u32x4*)(vp1 + lh * 16);
      fa1.q[1] = *(const u32x4*)(vp1 + lh * 16 + 8);
    }
    o0 = WMMA_BF16(fa0.v, bp, o0);
    o1 = WMMA_BF16(fa1.v, bp, o1);
  }

  float inv = 1.f / fmaxf(s_run, 1e-20f);
#pragma unroll
  for (int r = 0; r < 8; ++r) {
    int dh0 = r + 8 * lh;
    att[(size_t)(qb + q) * 512 + h * 32 + dh0]      = o0[r] * inv;
    att[(size_t)(qb + q) * 512 + h * 32 + 16 + dh0] = o1[r] * inv;
  }
}

// ------------------------------------------------------------- elementwise epilogues
__global__ void gate_add_kernel(const float* __restrict__ h0, const float* __restrict__ G,
                                const float* __restrict__ attO, const float* __restrict__ mask,
                                float* __restrict__ h1, int total, int D) {
  int i = blockIdx.x * blockDim.x + threadIdx.x;
  if (i < total) {
    float m = mask[i / D];
    float sg = 1.f / (1.f + __expf(-G[i]));
    h1[i] = h0[i] + sg * attO[i] * m;
  }
}

__global__ void swiglu_kernel(const float* __restrict__ t1, const float* __restrict__ t3,
                              unsigned short* __restrict__ out, int total) {
  int i = blockIdx.x * blockDim.x + threadIdx.x;
  if (i < total) {
    float a = t1[i];
    float s = a / (1.f + __expf(-a));
    out[i] = f32_to_bf16_bits(s * t3[i]);
  }
}

__global__ void add_mask_kernel(const float* __restrict__ h1, const float* __restrict__ ff,
                                const float* __restrict__ mask, float* __restrict__ h2,
                                int total, int D) {
  int i = blockIdx.x * blockDim.x + threadIdx.x;
  if (i < total) h2[i] = h1[i] + ff[i] * mask[i / D];
}

// ------------------------------------------------------------- Sinkhorn log-kernel (WMMA)
__global__ __launch_bounds__(32) void logk_kernel(
    const unsigned short* __restrict__ Qsh, const unsigned short* __restrict__ Ksh,
    const float* __restrict__ mask, const float* __restrict__ alpha_h,
    const float* __restrict__ r_h, const float* __restrict__ eps,
    float* __restrict__ logK, int N) {
  const int h = blockIdx.y;
  const int nb = blockIdx.x * 16;
  const int lane = threadIdx.x;
  const int q = lane & 15, lh = lane >> 4;

  Frag bq;
  {
    const unsigned short* Qp = Qsh + ((size_t)h * N + nb + q) * 32;
    bq.q[0] = *(const u32x4*)(Qp + lh * 8);
    bq.q[1] = *(const u32x4*)(Qp + 16 + lh * 8);
  }
  const float rh = r_h[h], ah = alpha_h[h], ie = 1.f / eps[h];
  const float maskn = mask[nb + q];

  for (int mb = 0; mb < N; mb += 16) {
    Frag ak;
    {
      const unsigned short* Kp = Ksh + ((size_t)h * N + mb + q) * 32;
      ak.q[0] = *(const u32x4*)(Kp + lh * 8);
      ak.q[1] = *(const u32x4*)(Kp + 16 + lh * 8);
    }
    v8f z = {};
    v8f s = WMMA_BF16(ak.v, bq.v, z);   // sim^T tile: M=m(r+8*lh), N=n(lane&15)
#pragma unroll
    for (int r = 0; r < 8; ++r) {
      int mg = mb + r + 8 * lh;
      float val = (rh * s[r] + ah) * ie;
      if (mask[mg] <= 0.f || maskn <= 0.f) val = NEGV;
      logK[((size_t)h * N + (nb + q)) * N + mg] = val;
    }
  }
}

// ------------------------------------------------------------- Sinkhorn iterations
__global__ __launch_bounds__(256) void nvalid_kernel(const float* __restrict__ mask,
                                                     float* __restrict__ out, int N) {
  __shared__ float red[256];
  float s = 0.f;
  for (int i = threadIdx.x; i < N; i += 256) s += mask[i];
  red[threadIdx.x] = s; __syncthreads();
  for (int st = 128; st > 0; st >>= 1) { if (threadIdx.x < st) red[threadIdx.x] += red[threadIdx.x + st]; __syncthreads(); }
  if (threadIdx.x == 0) out[0] = red[0];
}

__global__ __launch_bounds__(256) void sink_f_kernel(
    const float* __restrict__ logK, const float* __restrict__ gv,
    const float* __restrict__ mask, const float* __restrict__ nvalid,
    float* __restrict__ fv, int N) {
  const int h = blockIdx.y;
  const int wave = threadIdx.x >> 5, lane = threadIdx.x & 31;
  const int n = blockIdx.x * 8 + wave;
  const float* row = logK + ((size_t)h * N + n) * N;
  const float* gh = gv + (size_t)h * N;
  float mx = -3.0e38f, sm = 0.f;
  for (int m = lane; m < N; m += 32) {
    float v = row[m] + gh[m];
    float nm = fmaxf(mx, v);
    sm = sm * __expf(mx - nm) + __expf(v - nm);
    mx = nm;
  }
#pragma unroll
  for (int o = 16; o > 0; o >>= 1) {
    float omx = __shfl_xor(mx, o, 32), osm = __shfl_xor(sm, o, 32);
    float nm = fmaxf(mx, omx);
    sm = sm * __expf(mx - nm) + osm * __expf(omx - nm);
    mx = nm;
  }
  if (lane == 0) {
    float la = (mask[n] > 0.f) ? -__logf(nvalid[0]) : NEGV;
    fv[(size_t)h * N + n] = la - (mx + __logf(sm));
  }
}

__global__ __launch_bounds__(256) void sink_g_kernel(
    const float* __restrict__ logK, const float* __restrict__ fv,
    const float* __restrict__ mask, const float* __restrict__ nvalid,
    float* __restrict__ gv, int N) {
  const int h = blockIdx.y;
  const int wave = threadIdx.x >> 5, lane = threadIdx.x & 31;
  const int m = blockIdx.x * 8 + wave;
  const float* fh = fv + (size_t)h * N;
  float mx = -3.0e38f, sm = 0.f;
  for (int n = lane; n < N; n += 32) {
    float v = logK[((size_t)h * N + n) * N + m] + fh[n];
    float nm = fmaxf(mx, v);
    sm = sm * __expf(mx - nm) + __expf(v - nm);
    mx = nm;
  }
#pragma unroll
  for (int o = 16; o > 0; o >>= 1) {
    float omx = __shfl_xor(mx, o, 32), osm = __shfl_xor(sm, o, 32);
    float nm = fmaxf(mx, omx);
    sm = sm * __expf(mx - nm) + osm * __expf(omx - nm);
    mx = nm;
  }
  if (lane == 0) {
    float la = (mask[m] > 0.f) ? -__logf(nvalid[0]) : NEGV;
    gv[(size_t)h * N + m] = la - (mx + __logf(sm));
  }
}

// --------------------------------------------- transport apply: per-head weighted barycenter
__global__ __launch_bounds__(256) void transport_kernel(
    const float* __restrict__ logK, const float* __restrict__ fv, const float* __restrict__ gv,
    const float* __restrict__ x_res, const float* __restrict__ lambda_raw,
    float* __restrict__ xch, int N) {
  const int h = blockIdx.y;
  const int wave = threadIdx.x >> 5, lane = threadIdx.x & 31;
  const int n = blockIdx.x * 8 + wave;
  const float* row = logK + ((size_t)h * N + n) * N;
  const float* gh = gv + (size_t)h * N;
  const float fn = fv[(size_t)h * N + n];
  float mx = -3.0e38f, sm = 0.f, a0 = 0.f, a1 = 0.f, a2 = 0.f;
  for (int m = lane; m < N; m += 32) {
    float v = fn + row[m] + gh[m];
    float nm = fmaxf(mx, v);
    float sc = __expf(mx - nm);
    float e = __expf(v - nm);
    sm = sm * sc + e;
    a0 = a0 * sc + e * x_res[m * 3 + 0];
    a1 = a1 * sc + e * x_res[m * 3 + 1];
    a2 = a2 * sc + e * x_res[m * 3 + 2];
    mx = nm;
  }
#pragma unroll
  for (int o = 16; o > 0; o >>= 1) {
    float omx = __shfl_xor(mx, o, 32), osm = __shfl_xor(sm, o, 32);
    float oa0 = __shfl_xor(a0, o, 32), oa1 = __shfl_xor(a1, o, 32), oa2 = __shfl_xor(a2, o, 32);
    float nm = fmaxf(mx, omx);
    float s1 = __expf(mx - nm), s2 = __expf(omx - nm);
    sm = sm * s1 + osm * s2;
    a0 = a0 * s1 + oa0 * s2;
    a1 = a1 * s1 + oa1 * s2;
    a2 = a2 * s1 + oa2 * s2;
    mx = nm;
  }
  if (lane == 0) {
    float inv = 1.f / fmaxf(sm, 1e-30f);
    float lam = tanhf(lambda_raw[h]) / 16.f;
    size_t base = ((size_t)h * N + n) * 3;
    xch[base + 0] = lam * (a0 * inv - x_res[n * 3 + 0]);
    xch[base + 1] = lam * (a1 * inv - x_res[n * 3 + 1]);
    xch[base + 2] = lam * (a2 * inv - x_res[n * 3 + 2]);
  }
}

__global__ void disp_reduce_kernel(const float* __restrict__ xch, float* __restrict__ disp,
                                   int total, int N) {
  int i = blockIdx.x * blockDim.x + threadIdx.x;
  if (i < total) {
    float s = 0.f;
    for (int h = 0; h < 16; ++h) s += xch[(size_t)h * N * 3 + i];
    disp[i] = s;
  }
}

// --------------------------------------------- final gate + x_out
__global__ __launch_bounds__(32) void xout_kernel(
    const float* __restrict__ h2, const float* __restrict__ lg, const float* __restrict__ lb,
    const float* __restrict__ wgw, const float* __restrict__ wgb,
    const float* __restrict__ x_res, const float* __restrict__ disp,
    const float* __restrict__ mask, float* __restrict__ x_out, int D) {
  int n = blockIdx.x, lane = threadIdx.x;
  const float* xr = h2 + (size_t)n * D;
  float s = 0.f, s2 = 0.f;
  for (int d = lane; d < D; d += 32) { float v = xr[d]; s += v; s2 += v * v; }
#pragma unroll
  for (int o = 16; o > 0; o >>= 1) { s += __shfl_xor(s, o, 32); s2 += __shfl_xor(s2, o, 32); }
  float mu = s / (float)D;
  float var = s2 / (float)D - mu * mu;
  float rs = rsqrtf(var + 1e-5f);
  float dot = 0.f;
  for (int d = lane; d < D; d += 32) dot += ((xr[d] - mu) * rs * lg[d] + lb[d]) * wgw[d];
#pragma unroll
  for (int o = 16; o > 0; o >>= 1) dot += __shfl_xor(dot, o, 32);
  float gate = 1.f / (1.f + __expf(-(dot + wgb[0])));
  if (lane < 3)
    x_out[n * 3 + lane] = x_res[n * 3 + lane] + gate * disp[n * 3 + lane] * mask[n];
}

// ================================================================ host driver
extern "C" void kernel_launch(void* const* d_in, const int* in_sizes, int n_in,
                              void* d_out, int out_size, void* d_ws, size_t ws_size,
                              hipStream_t stream) {
  (void)in_sizes; (void)n_in; (void)out_size; (void)ws_size;
  const float* h0         = (const float*)d_in[0];
  const float* x_res      = (const float*)d_in[1];
  const int*   pos_bins   = (const int*)d_in[2];
  const float* mask       = (const float*)d_in[3];
  const float* ln_mha_g   = (const float*)d_in[4];
  const float* ln_mha_b   = (const float*)d_in[5];
  const float* w_q        = (const float*)d_in[6];
  const float* w_k        = (const float*)d_in[7];
  const float* w_v        = (const float*)d_in[8];
  const float* w_g        = (const float*)d_in[9];
  const float* w_o        = (const float*)d_in[10];
  const float* pos_w      = (const float*)d_in[11];
  const float* ln_ff_g    = (const float*)d_in[12];
  const float* ln_ff_b    = (const float*)d_in[13];
  const float* sw_w1      = (const float*)d_in[14];
  const float* sw_w3      = (const float*)d_in[15];
  const float* sw_w2      = (const float*)d_in[16];
  const float* ln_sink_g  = (const float*)d_in[17];
  const float* ln_sink_b  = (const float*)d_in[18];
  const float* w_q_sink   = (const float*)d_in[19];
  const float* w_k_sink   = (const float*)d_in[20];
  const float* alpha_h    = (const float*)d_in[21];
  const float* r_h        = (const float*)d_in[22];
  const float* eps        = (const float*)d_in[23];
  const float* ln_gate_g  = (const float*)d_in[24];
  const float* ln_gate_b  = (const float*)d_in[25];
  const float* w_gate_w   = (const float*)d_in[26];
  const float* w_gate_b   = (const float*)d_in[27];
  const float* lambda_raw = (const float*)d_in[28];

  constexpr int N = 1024, D = 512, H = 16, DH = 32, FF = 2048, KIT = 20;

  char* ws = (char*)d_ws;
  size_t off = 0;
  auto alloc = [&](size_t bytes) -> char* {
    char* p = ws + off;
    off += (bytes + 255) & ~(size_t)255;
    return p;
  };

  // all weights stored transposed (Nw x K) in bf16
  unsigned short* wq_b  = (unsigned short*)alloc((size_t)D * D * 2);
  unsigned short* wk_b  = (unsigned short*)alloc((size_t)D * D * 2);
  unsigned short* wv_b  = (unsigned short*)alloc((size_t)D * D * 2);
  unsigned short* wg_b  = (unsigned short*)alloc((size_t)D * D * 2);
  unsigned short* wo_b  = (unsigned short*)alloc((size_t)D * D * 2);
  unsigned short* w1_b  = (unsigned short*)alloc((size_t)D * FF * 2);
  unsigned short* w3_b  = (unsigned short*)alloc((size_t)D * FF * 2);
  unsigned short* w2_b  = (unsigned short*)alloc((size_t)FF * D * 2);
  unsigned short* wqs_b = (unsigned short*)alloc((size_t)D * D * 2);
  unsigned short* wks_b = (unsigned short*)alloc((size_t)D * D * 2);

  unsigned short* hn_b = (unsigned short*)alloc((size_t)N * D * 2);
  float* Cq  = (float*)alloc((size_t)N * D * 4);
  float* Ck  = (float*)alloc((size_t)N * D * 4);
  float* Cv  = (float*)alloc((size_t)N * D * 4);
  float* Cg  = (float*)alloc((size_t)N * D * 4);
  unsigned short* Qh = (unsigned short*)alloc((size_t)H * N * DH * 2);
  unsigned short* Kh = (unsigned short*)alloc((size_t)H * N * DH * 2);
  unsigned short* Vt = (unsigned short*)alloc((size_t)H * DH * N * 2);
  float* att  = (float*)alloc((size_t)N * D * 4);
  unsigned short* att_b = (unsigned short*)alloc((size_t)N * D * 2);
  float* attO = (float*)alloc((size_t)N * D * 4);
  float* h1   = (float*)alloc((size_t)N * D * 4);
  unsigned short* hf_b = (unsigned short*)alloc((size_t)N * D * 2);
  float* t1   = (float*)alloc((size_t)N * FF * 4);
  float* t3   = (float*)alloc((size_t)N * FF * 4);
  unsigned short* ffb = (unsigned short*)alloc((size_t)N * FF * 2);
  float* ffO  = (float*)alloc((size_t)N * D * 4);
  float* h2   = (float*)alloc((size_t)N * D * 4);
  unsigned short* hs_b = (unsigned short*)alloc((size_t)N * D * 2);
  float* Cqs  = (float*)alloc((size_t)N * D * 4);
  float* Cks  = (float*)alloc((size_t)N * D * 4);
  unsigned short* Qsh = (unsigned short*)alloc((size_t)H * N * DH * 2);
  unsigned short* Ksh = (unsigned short*)alloc((size_t)H * N * DH * 2);
  float* logK = (float*)alloc((size_t)H * N * N * 4);   // 64 MB -> L2 resident
  float* fv   = (float*)alloc((size_t)H * N * 4);
  float* gv   = (float*)alloc((size_t)H * N * 4);
  float* nv   = (float*)alloc(256);
  float* xch  = (float*)alloc((size_t)H * N * 3 * 4);
  float* disp = (float*)alloc((size_t)N * 3 * 4);

  // --- weight conversion to transposed bf16
  auto cvtT = [&](const float* s, unsigned short* d, int K, int Nw) {
    dim3 g(Nw / 32, K / 32);
    cvt_t_kernel<<<g, 256, 0, stream>>>(s, d, K, Nw);
  };
  cvtT(w_q, wq_b, D, D);  cvtT(w_k, wk_b, D, D);  cvtT(w_v, wv_b, D, D);
  cvtT(w_g, wg_b, D, D);  cvtT(w_o, wo_b, D, D);
  cvtT(sw_w1, w1_b, D, FF); cvtT(sw_w3, w3_b, D, FF); cvtT(sw_w2, w2_b, FF, D);
  cvtT(w_q_sink, wqs_b, D, D); cvtT(w_k_sink, wks_b, D, D);

  dim3 blk128(128), blk256(256), blk32(32);
  dim3 g8(D / 64, N / 64);      // Nw=512 GEMMs
  dim3 g32(FF / 64, N / 64);    // Nw=2048 GEMMs
  dim3 grp(N, H);
  dim3 ga(N / 16, H);
  dim3 gs(N / 8, H);

  // --- MHA
  ln_bf16_kernel<<<N, blk256, 0, stream>>>(h0, ln_mha_g, ln_mha_b, hn_b, D);
  gemm_bf16_kernel<<<g8, blk128, 0, stream>>>(hn_b, wq_b, Cq, N, D, D);
  gemm_bf16_kernel<<<g8, blk128, 0, stream>>>(hn_b, wk_b, Ck, N, D, D);
  gemm_bf16_kernel<<<g8, blk128, 0, stream>>>(hn_b, wv_b, Cv, N, D, D);
  gemm_bf16_kernel<<<g8, blk128, 0, stream>>>(hn_b, wg_b, Cg, N, D, D);
  repack_kernel<<<grp, blk32, 0, stream>>>(Cq, Qh, N, 0);
  repack_kernel<<<grp, blk32, 0, stream>>>(Ck, Kh, N, 0);
  repack_kernel<<<grp, blk32, 0, stream>>>(Cv, Vt, N, 2);   // transposed (H,DH,N)
  attn_kernel<<<ga, blk32, 0, stream>>>(Qh, Kh, Vt, pos_bins, pos_w, mask, att, N);
  cvt_kernel<<<(N * D + 255) / 256, blk256, 0, stream>>>(att, att_b, N * D);
  gemm_bf16_kernel<<<g8, blk128, 0, stream>>>(att_b, wo_b, attO, N, D, D);
  gate_add_kernel<<<(N * D + 255) / 256, blk256, 0, stream>>>(h0, Cg, attO, mask, h1, N * D, D);

  // --- SwiGLU FFN
  ln_bf16_kernel<<<N, blk256, 0, stream>>>(h1, ln_ff_g, ln_ff_b, hf_b, D);
  gemm_bf16_kernel<<<g32, blk128, 0, stream>>>(hf_b, w1_b, t1, N, FF, D);
  gemm_bf16_kernel<<<g32, blk128, 0, stream>>>(hf_b, w3_b, t3, N, FF, D);
  swiglu_kernel<<<(N * FF + 255) / 256, blk256, 0, stream>>>(t1, t3, ffb, N * FF);
  gemm_bf16_kernel<<<g8, blk128, 0, stream>>>(ffb, w2_b, ffO, N, D, FF);
  add_mask_kernel<<<(N * D + 255) / 256, blk256, 0, stream>>>(h1, ffO, mask, h2, N * D, D);

  // --- Sinkhorn OT
  ln_bf16_kernel<<<N, blk256, 0, stream>>>(h2, ln_sink_g, ln_sink_b, hs_b, D);
  gemm_bf16_kernel<<<g8, blk128, 0, stream>>>(hs_b, wqs_b, Cqs, N, D, D);
  gemm_bf16_kernel<<<g8, blk128, 0, stream>>>(hs_b, wks_b, Cks, N, D, D);
  repack_kernel<<<grp, blk32, 0, stream>>>(Cqs, Qsh, N, 1);
  repack_kernel<<<grp, blk32, 0, stream>>>(Cks, Ksh, N, 1);
  logk_kernel<<<ga, blk32, 0, stream>>>(Qsh, Ksh, mask, alpha_h, r_h, eps, logK, N);
  nvalid_kernel<<<1, blk256, 0, stream>>>(mask, nv, N);
  zero_kernel<<<(H * N + 255) / 256, blk256, 0, stream>>>(fv, H * N);
  zero_kernel<<<(H * N + 255) / 256, blk256, 0, stream>>>(gv, H * N);
  for (int it = 0; it < KIT; ++it) {
    sink_f_kernel<<<gs, blk256, 0, stream>>>(logK, gv, mask, nv, fv, N);
    sink_g_kernel<<<gs, blk256, 0, stream>>>(logK, fv, mask, nv, gv, N);
  }
  transport_kernel<<<gs, blk256, 0, stream>>>(logK, fv, gv, x_res, lambda_raw, xch, N);
  disp_reduce_kernel<<<(N * 3 + 255) / 256, blk256, 0, stream>>>(xch, disp, N * 3, N);

  // --- outputs: h (N*D) then x_out (N*3)
  copy_kernel<<<(N * D + 255) / 256, blk256, 0, stream>>>(h2, (float*)d_out, N * D);
  xout_kernel<<<N, blk32, 0, stream>>>(h2, ln_gate_g, ln_gate_b, w_gate_w, w_gate_b,
                                       x_res, disp, mask, (float*)d_out + (size_t)N * D, D);
}
